// MidLevelRNN_6932077216195
// MI455X (gfx1250) — compile-verified, matrix operand order checked by reading, not run on hardware
//
#include <hip/hip_runtime.h>
#include <hip/hip_bf16.h>
#include <math.h>

#define X_DIM   2048
#define H_DIM   4096
#define Y_DIM   2048
#define ROLLOUT 1024
#define XH_DIM  (X_DIM + H_DIM)

typedef __attribute__((ext_vector_type(16))) __bf16 v16bf;
typedef __attribute__((ext_vector_type(8)))  float  v8f;

// ---------- helpers ----------
__device__ __forceinline__ unsigned short f2bf(float f) {
    unsigned int u = __float_as_uint(f);
    u += 0x7FFFu + ((u >> 16) & 1u);      // round to nearest even
    return (unsigned short)(u >> 16);
}
__device__ __forceinline__ float bflo(unsigned int u) { return __uint_as_float(u << 16); }
__device__ __forceinline__ float bfhi(unsigned int u) { return __uint_as_float(u & 0xFFFF0000u); }

// ---------- weight split + fp32 -> bf16 conversion ----------
// W_xh is [H_DIM, XH_DIM] row-major; split into Wx [H_DIM, X_DIM] and Whh [H_DIM, H_DIM] (bf16)
__global__ void split_wxh_kernel(const float* __restrict__ Wxh,
                                 unsigned short* __restrict__ Wx,
                                 unsigned short* __restrict__ Whh) {
    const long long total = (long long)H_DIM * XH_DIM;
    for (long long idx = (long long)blockIdx.x * blockDim.x + threadIdx.x;
         idx < total; idx += (long long)gridDim.x * blockDim.x) {
        int row = (int)(idx / XH_DIM);
        int col = (int)(idx % XH_DIM);
        unsigned short v = f2bf(Wxh[idx]);
        if (col < X_DIM) Wx [(long long)row * X_DIM + col]          = v;
        else             Whh[(long long)row * H_DIM + (col - X_DIM)] = v;
    }
}

__global__ void cvt_bf16_kernel(const float* __restrict__ src,
                                unsigned short* __restrict__ dst, long long n) {
    for (long long i = (long long)blockIdx.x * blockDim.x + threadIdx.x;
         i < n; i += (long long)gridDim.x * blockDim.x) {
        dst[i] = f2bf(src[i]);
    }
}

// ---------- WMMA GEMM:  C[M,N] = A[M,K] * B[N,K]^T + bias[N] ----------
// A, B bf16 row-major; C fp32 row-major. Requires M%64==0, N%64==0, K%32==0.
// Block = 128 threads (4 waves); wave computes a 16(M) x 64(N) strip (4 WMMA tiles, A reused 4x).
__global__ void wmma_gemm_bt_kernel(const unsigned short* __restrict__ A,
                                    const unsigned short* __restrict__ B,
                                    const float* __restrict__ bias,
                                    float* __restrict__ C,
                                    int M, int N, int K) {
    const int lane = threadIdx.x & 31;
    const int wave = threadIdx.x >> 5;          // 0..3
    const int m0   = blockIdx.y * 64 + wave * 16;
    const int n0   = blockIdx.x * 64;
    const int r16  = lane & 15;                 // row-within-tile index (M for A, N for B)
    const int half = lane >> 4;                 // selects K sub-half (16 elems)

    v8f acc[4];
#pragma unroll
    for (int j = 0; j < 4; ++j)
#pragma unroll
        for (int r = 0; r < 8; ++r) acc[j][r] = 0.0f;

    const unsigned short* arow = A + (long long)(m0 + r16) * K + 16 * half;
    const unsigned short* brow = B + (long long)(n0 + r16) * K + 16 * half;
    const long long bstep = (long long)16 * K;  // next 16-row B tile

    for (int k0 = 0; k0 < K; k0 += 32) {
        v16bf a = *reinterpret_cast<const v16bf*>(arow + k0);
#pragma unroll
        for (int j = 0; j < 4; ++j) {
            v16bf b = *reinterpret_cast<const v16bf*>(brow + (long long)j * bstep + k0);
            acc[j] = __builtin_amdgcn_wmma_f32_16x16x32_bf16(
                         false, a, false, b, (short)0, acc[j], false, false);
        }
    }

    // C/D layout: lanes 0-15 -> N=lane, M=r ; lanes 16-31 -> N=lane-16, M=r+8
#pragma unroll
    for (int j = 0; j < 4; ++j) {
        int n = n0 + j * 16 + r16;
        float bv = bias[n];
#pragma unroll
        for (int r = 0; r < 8; ++r) {
            int m = m0 + r + 8 * half;
            C[(long long)m * N + n] = acc[j][r] + bv;
        }
    }
}

// ---------- one recurrence step: h_{t+1} = tanh(Whh * h_t + Xp[t]) ----------
// Whh bf16 [H_DIM,H_DIM] (L2-resident, 32MB). Wave-per-output-row, 8 waves/block.
__global__ void rnn_step_kernel(const unsigned short* __restrict__ Whh,
                                const float* __restrict__ Xp,
                                unsigned short* __restrict__ Hbf,
                                float* __restrict__ hfinal,
                                int t) {
    const int lane = threadIdx.x & 31;
    const int wave = threadIdx.x >> 5;
    const int row  = blockIdx.x * 8 + wave;      // 512 blocks * 8 waves = 4096 rows

    const unsigned short* hprev = Hbf + (long long)t * H_DIM;
    const unsigned short* w     = Whh + (long long)row * H_DIM;

    float s = 0.0f;
#pragma unroll
    for (int c = 0; c < H_DIM / 256; ++c) {      // 16 chunks of 8 bf16 per lane
        int k = c * 256 + lane * 8;
        uint4 wu = *reinterpret_cast<const uint4*>(w + k);
        uint4 hu = *reinterpret_cast<const uint4*>(hprev + k);
        s = fmaf(bflo(wu.x), bflo(hu.x), s);
        s = fmaf(bfhi(wu.x), bfhi(hu.x), s);
        s = fmaf(bflo(wu.y), bflo(hu.y), s);
        s = fmaf(bfhi(wu.y), bfhi(hu.y), s);
        s = fmaf(bflo(wu.z), bflo(hu.z), s);
        s = fmaf(bfhi(wu.z), bfhi(hu.z), s);
        s = fmaf(bflo(wu.w), bflo(hu.w), s);
        s = fmaf(bfhi(wu.w), bfhi(hu.w), s);
    }
#pragma unroll
    for (int off = 16; off > 0; off >>= 1) s += __shfl_xor(s, off, 32);

    if (lane == 0) {
        float v = tanhf(s + Xp[(long long)t * H_DIM + row]);
        Hbf[(long long)(t + 1) * H_DIM + row] = f2bf(v);
        if (t == ROLLOUT - 1) hfinal[row] = v;   // exact fp32 final state
    }
}

// ---------- host-side launch ----------
extern "C" void kernel_launch(void* const* d_in, const int* in_sizes, int n_in,
                              void* d_out, int out_size, void* d_ws, size_t ws_size,
                              hipStream_t stream) {
    const float* x    = (const float*)d_in[0];   // [1, 1024, 2048]
    const float* h0   = (const float*)d_in[1];   // [4096]
    const float* W_xh = (const float*)d_in[2];   // [4096, 6144]
    const float* b_xh = (const float*)d_in[3];   // [4096]
    const float* W_hy = (const float*)d_in[4];   // [2048, 4096]
    const float* b_hy = (const float*)d_in[5];   // [2048]
    float* out = (float*)d_out;                  // outputs [1024*2048] then h_final [4096]

    char* ws = (char*)d_ws;
    // workspace layout (bytes)
    const size_t OFF_XP  = 0;                                   // fp32 [1024,4096]  16 MB
    const size_t OFF_HBF = OFF_XP  + (size_t)ROLLOUT * H_DIM * 4;   // bf16 [1025,4096]
    const size_t OFF_XBF = OFF_HBF + (size_t)(ROLLOUT + 1) * H_DIM * 2;
    const size_t OFF_WX  = OFF_XBF + (size_t)ROLLOUT * X_DIM * 2;
    const size_t OFF_WHH = OFF_WX  + (size_t)H_DIM * X_DIM * 2;
    const size_t OFF_WHY = OFF_WHH + (size_t)H_DIM * H_DIM * 2;

    float*          Xp     = (float*)         (ws + OFF_XP);
    unsigned short* Hbf    = (unsigned short*)(ws + OFF_HBF);
    unsigned short* x_bf   = (unsigned short*)(ws + OFF_XBF);
    unsigned short* Wx_bf  = (unsigned short*)(ws + OFF_WX);
    unsigned short* Whh_bf = (unsigned short*)(ws + OFF_WHH);
    unsigned short* Why_bf = (unsigned short*)(ws + OFF_WHY);

    // 1) convert weights / inputs to bf16 (recomputed every call -> deterministic)
    split_wxh_kernel<<<4096, 256, 0, stream>>>(W_xh, Wx_bf, Whh_bf);
    cvt_bf16_kernel<<<2048, 256, 0, stream>>>(x,    x_bf,   (long long)ROLLOUT * X_DIM);
    cvt_bf16_kernel<<<2048, 256, 0, stream>>>(W_hy, Why_bf, (long long)Y_DIM * H_DIM);
    cvt_bf16_kernel<<<16,   256, 0, stream>>>(h0,   Hbf,    (long long)H_DIM);   // row 0 = h_0

    // 2) batched input projection: Xp = x @ Wx^T + b_xh   (WMMA)
    {
        dim3 grid(H_DIM / 64, ROLLOUT / 64);   // (64, 16)
        wmma_gemm_bt_kernel<<<grid, 128, 0, stream>>>(x_bf, Wx_bf, b_xh, Xp,
                                                      ROLLOUT, H_DIM, X_DIM);
    }

    // 3) sequential recurrence: one graph node per step (grid-wide sync via launch order)
    for (int t = 0; t < ROLLOUT; ++t) {
        rnn_step_kernel<<<H_DIM / 8, 256, 0, stream>>>(
            Whh_bf, Xp, Hbf, out + (size_t)ROLLOUT * Y_DIM, t);
    }

    // 4) batched output projection: Y = H @ W_hy^T + b_hy   (WMMA) -> d_out[0 : 1024*2048)
    {
        dim3 grid(Y_DIM / 64, ROLLOUT / 64);   // (32, 16)
        wmma_gemm_bt_kernel<<<grid, 128, 0, stream>>>(Hbf + H_DIM, Why_bf, b_hy, out,
                                                      ROLLOUT, Y_DIM, H_DIM);
    }
}
